// Transducer_85753317032619
// MI455X (gfx1250) — matrix operand order repaired and checked by reading, not compiled, not run on hardware
//
#include <hip/hip_runtime.h>
#include <hip/hip_bf16.h>

typedef __attribute__((ext_vector_type(16))) _Float16 v16h;
typedef __attribute__((ext_vector_type(8)))  _Float16 v8h;
typedef __attribute__((ext_vector_type(8)))  float    v8f;

#define B_    8
#define T_    200
#define TPAD  208
#define U_    50
#define U1    51
#define U1P   52
#define H_    512
#define INNER 512
#define V_    1024

// ---------------- workspace layout (bytes, 256-aligned) ----------------
#define OFF_ENCP   0              // 8*208*512 f32 = 3,407,872
#define OFF_DECP   3407872        // 8*52*512  f32 =   851,968
#define OFF_W2H    4259840        // 1024*512  f16 = 1,048,576
#define OFF_BLANK  5308416        // 8*200*51  f32 =   326,400
#define OFF_EMIT   5634816        // 8*200*51  f32 =   326,400
#define OFF_LL     5961216        // 8 f32

// ---------------- W2 f32 -> f16 ----------------
__global__ __launch_bounds__(256) void convw2_kernel(const float* __restrict__ W2,
                                                     _Float16* __restrict__ w2h) {
  int idx = blockIdx.x * 256 + threadIdx.x;   // 1024*512 = 524288 total
  w2h[idx] = (_Float16)W2[idx];
}

// ---------------- enc projection: encp[b,t,i] = sum_h enc[b,t,h]*W1[i,h] ----------------
__global__ __launch_bounds__(256) void encproj_kernel(const float* __restrict__ enc,
                                                      const float* __restrict__ W1,
                                                      float* __restrict__ out) {
  size_t idx = (size_t)blockIdx.x * 256 + threadIdx.x;  // 8*208*512
  int i   = (int)(idx & 511);
  int row = (int)(idx >> 9);          // b*208 + t
  int t = row % TPAD, b = row / TPAD;
  float s = 0.f;
  if (t < T_) {
    const float* e = enc + ((size_t)b * T_ + t) * H_;
    const float* w = W1 + (size_t)i * (2 * H_);
#pragma unroll 4
    for (int h = 0; h < H_; ++h) s += e[h] * w[h];
  }
  out[idx] = s;
}

// ---------------- dec projection (+b1): decp[b,u,i] = b1[i] + sum_h dec[b,u,h]*W1[i,H+h] ----
__global__ __launch_bounds__(256) void decproj_kernel(const float* __restrict__ dec,
                                                      const float* __restrict__ W1,
                                                      const float* __restrict__ b1,
                                                      float* __restrict__ out) {
  size_t idx = (size_t)blockIdx.x * 256 + threadIdx.x;  // 8*52*512
  int i   = (int)(idx & 511);
  int row = (int)(idx >> 9);          // b*52 + u
  int u = row % U1P, b = row / U1P;
  float s = 0.f;
  if (u < U1) {
    const float* d = dec + ((size_t)b * U1 + u) * H_;
    const float* w = W1 + (size_t)i * (2 * H_) + H_;
    s = b1[i];
#pragma unroll 4
    for (int h = 0; h < H_; ++h) s += d[h] * w[h];
  }
  out[idx] = s;
}

// ---------------- fused joint + logits GEMM (WMMA f16) + log-softmax stats ----------------
// grid (13, 13, 8) = (t-tile, u-tile, b); block 512 = 16 waves.
// Block tile: M=64 positions (16 t-rows x 4 u-values) x all V=1024 logits, kept in registers.
// Each wave: 4 N-tiles x 4 M-tiles -> each B fragment feeds 4 WMMAs (4x L2 traffic cut).
#define LDS_A_STRIDE 520   // 512 + 8 halves: 260 dwords % 64 banks = 4 -> conflict-free

__global__ __launch_bounds__(512) void joint_gemm_kernel(
    const float*    __restrict__ encp,      // [B,208,512]
    const float*    __restrict__ decp,      // [B,52,512] (b1 folded, u=51 zero pad)
    const _Float16* __restrict__ w2h,       // [1024,512] row-major
    const float*    __restrict__ b2,        // [1024]
    const int*      __restrict__ targets,   // [B,50]
    float* __restrict__ blank_lp,           // [B,200,51]
    float* __restrict__ emit_lp) {          // [B,200,51] (u<50 used)
  const int t0 = blockIdx.x * 16;
  const int u0 = blockIdx.y * 4;
  const int b  = blockIdx.z;
  const int tid  = threadIdx.x;
  const int lane = tid & 31, wave = tid >> 5;
  const int col  = lane & 15, half = lane >> 4;

  __shared__ _Float16 Ah[64 * LDS_A_STRIDE];   // 66,560 B
  __shared__ float red[16][64][2];             //  8,192 B
  __shared__ float blankv[64];
  __shared__ float emitv[64];

  // ---- build A tile: h = tanh(enc_proj + dec_proj+b1), f32 -> f16 in LDS ----
  // global row g = mt*16 + r  ->  (t = t0+r, u = u0+mt)
  const float* encB = encp + ((size_t)b * TPAD + t0) * INNER;
  const float* decB = decp + ((size_t)b * U1P + u0) * INNER;
#pragma unroll
  for (int it = 0; it < 64; ++it) {
    int idx = tid + it * 512;          // 64*512 = 32768
    int g = idx >> 9, c = idx & 511;
    int mt = g >> 4, r = g & 15;
    float hv = tanhf(encB[(size_t)r * INNER + c] + decB[(size_t)mt * INNER + c]);
    Ah[g * LDS_A_STRIDE + c] = (_Float16)hv;
  }
  __syncthreads();

  // ---- GEMM: C[64 x 1024] = A[64 x 512] * W2^T ----
  v8f acc[4][4] = {};   // [ntile][mtile]
  for (int ks = 0; ks < 16; ++ks) {
    const int k0 = ks * 32;
    v16h afrag[4];
#pragma unroll
    for (int mt = 0; mt < 4; ++mt) {
      const _Float16* Ab = &Ah[(mt * 16 + col) * LDS_A_STRIDE + half * 16 + k0];
      v8h a0 = *(const v8h*)(Ab);
      v8h a1 = *(const v8h*)(Ab + 8);
      afrag[mt] = __builtin_shufflevector(a0, a1, 0, 1, 2, 3, 4, 5, 6, 7,
                                                  8, 9, 10, 11, 12, 13, 14, 15);
    }
#pragma unroll
    for (int n = 0; n < 4; ++n) {
      const int vcol = (wave * 4 + n) * 16 + col;       // lane's W2 row = B column
      v16h bfrag = *(const v16h*)(w2h + (size_t)vcol * INNER + k0 + half * 16);
#pragma unroll
      for (int mt = 0; mt < 4; ++mt)
        acc[n][mt] = __builtin_amdgcn_wmma_f32_16x16x32_f16(
            false, afrag[mt], false, bfrag, (short)0, acc[n][mt], false, false);
    }
  }

  // ---- add b2, capture blank(v=0) / emit(v=target[u]) logits ----
  int v_t[4];
#pragma unroll
  for (int mt = 0; mt < 4; ++mt) {
    const int u = u0 + mt;
    v_t[mt] = (u < U_) ? targets[b * U_ + u] : -1;
  }
#pragma unroll
  for (int n = 0; n < 4; ++n) {
    const int vcol = (wave * 4 + n) * 16 + col;
    const float bb = b2[vcol];
#pragma unroll
    for (int mt = 0; mt < 4; ++mt) {
#pragma unroll
      for (int j = 0; j < 8; ++j) acc[n][mt][j] += bb;
      if (vcol == 0)
        for (int j = 0; j < 8; ++j) blankv[mt * 16 + half * 8 + j] = acc[n][mt][j];
      if (vcol == v_t[mt])
        for (int j = 0; j < 8; ++j) emitv[mt * 16 + half * 8 + j] = acc[n][mt][j];
    }
  }

  // ---- wave-local (max, sumexp) per row: pass 1 max, pass 2 sum vs common max ----
  float rmax[4][8];
#pragma unroll
  for (int mt = 0; mt < 4; ++mt)
#pragma unroll
    for (int j = 0; j < 8; ++j) {
      float m = acc[0][mt][j];
#pragma unroll
      for (int n = 1; n < 4; ++n) m = fmaxf(m, acc[n][mt][j]);
      rmax[mt][j] = m;
    }
#pragma unroll
  for (int off = 1; off < 16; off <<= 1)
#pragma unroll
    for (int mt = 0; mt < 4; ++mt)
#pragma unroll
      for (int j = 0; j < 8; ++j)
        rmax[mt][j] = fmaxf(rmax[mt][j], __shfl_xor(rmax[mt][j], off, 32));

  float rsum[4][8];
#pragma unroll
  for (int mt = 0; mt < 4; ++mt)
#pragma unroll
    for (int j = 0; j < 8; ++j) {
      float s = 0.f;
#pragma unroll
      for (int n = 0; n < 4; ++n) s += __expf(acc[n][mt][j] - rmax[mt][j]);
      rsum[mt][j] = s;
    }
#pragma unroll
  for (int off = 1; off < 16; off <<= 1)
#pragma unroll
    for (int mt = 0; mt < 4; ++mt)
#pragma unroll
      for (int j = 0; j < 8; ++j)
        rsum[mt][j] += __shfl_xor(rsum[mt][j], off, 32);

  if (col == 0) {
#pragma unroll
    for (int mt = 0; mt < 4; ++mt)
#pragma unroll
      for (int j = 0; j < 8; ++j) {
        const int row = mt * 16 + half * 8 + j;
        red[wave][row][0] = rmax[mt][j];
        red[wave][row][1] = rsum[mt][j];
      }
  }
  __syncthreads();

  // ---- cross-wave combine (rescaled) + write two log-probs per row ----
  if (tid < 64) {
    const int g = tid;
    float m = red[0][g][0], s = red[0][g][1];
#pragma unroll
    for (int w = 1; w < 16; ++w) {
      float om = red[w][g][0], os = red[w][g][1];
      float nm = fmaxf(m, om);
      s = s * __expf(m - nm) + os * __expf(om - nm);
      m = nm;
    }
    const float lse = m + __logf(s);
    const int t = t0 + (g & 15);
    const int u = u0 + (g >> 4);
    if (t < T_ && u < U1) {
      const size_t o = ((size_t)b * T_ + t) * U1 + u;
      blank_lp[o] = blankv[g] - lse;
      if (u < U_) emit_lp[o] = emitv[g] - lse;
    }
  }
}

// ---------------- RNNT forward variable: anti-diagonal wavefront, 1 block / batch ----------
__global__ __launch_bounds__(64) void alpha_kernel(const float* __restrict__ blank_lp,
                                                   const float* __restrict__ emit_lp,
                                                   const int* __restrict__ inputs_length,
                                                   const int* __restrict__ targets_length,
                                                   float* __restrict__ ll) {
  const int b = blockIdx.x;
  const int u = threadIdx.x;
  __shared__ float prev[U1];
  const int t_idx = inputs_length[b] - 1;
  const int u_idx = targets_length[b];
  const float* blp = blank_lp + (size_t)b * T_ * U1;
  const float* elp = emit_lp + (size_t)b * T_ * U1;

  for (int d = 0; d < T_ + U1 - 1; ++d) {      // 250 diagonals
    float val = 0.f;
    bool valid = false;
    const int t = d - u;
    if (u <= U_ && t >= 0 && t < T_) {
      valid = true;
      if (t == 0 && u == 0) {
        val = 0.f;
      } else if (t == 0) {
        val = prev[u - 1] + elp[(size_t)0 * U1 + (u - 1)];
      } else if (u == 0) {
        val = prev[0] + blp[(size_t)(t - 1) * U1];
      } else {
        float a = prev[u] + blp[(size_t)(t - 1) * U1 + u];
        float c = prev[u - 1] + elp[(size_t)t * U1 + (u - 1)];
        float m = fmaxf(a, c);
        val = m + __logf(__expf(a - m) + __expf(c - m));
      }
      if (t == t_idx && u == u_idx) ll[b] = val + blp[(size_t)t * U1 + u];
    }
    __syncthreads();
    if (valid) prev[u] = val;
    __syncthreads();
  }
}

__global__ void finish_kernel(const float* __restrict__ ll, float* __restrict__ out) {
  if (threadIdx.x == 0) {
    float s = 0.f;
    for (int b = 0; b < B_; ++b) s += ll[b];
    out[0] = -s / (float)B_;
  }
}

extern "C" void kernel_launch(void* const* d_in, const int* in_sizes, int n_in,
                              void* d_out, int out_size, void* d_ws, size_t ws_size,
                              hipStream_t stream) {
  const float* enc = (const float*)d_in[0];
  const float* dec = (const float*)d_in[1];
  const float* W1  = (const float*)d_in[2];
  const float* b1  = (const float*)d_in[3];
  const float* W2  = (const float*)d_in[4];
  const float* b2  = (const float*)d_in[5];
  const int* targets = (const int*)d_in[6];
  const int* in_len  = (const int*)d_in[7];
  const int* tg_len  = (const int*)d_in[8];

  char* ws = (char*)d_ws;
  float*    encp     = (float*)(ws + OFF_ENCP);
  float*    decp     = (float*)(ws + OFF_DECP);
  _Float16* w2h      = (_Float16*)(ws + OFF_W2H);
  float*    blank_lp = (float*)(ws + OFF_BLANK);
  float*    emit_lp  = (float*)(ws + OFF_EMIT);
  float*    ll       = (float*)(ws + OFF_LL);

  convw2_kernel<<<(V_ * INNER) / 256, 256, 0, stream>>>(W2, w2h);
  encproj_kernel<<<(B_ * TPAD * INNER) / 256, 256, 0, stream>>>(enc, W1, encp);
  decproj_kernel<<<(B_ * U1P * INNER) / 256, 256, 0, stream>>>(dec, W1, b1, decp);
  joint_gemm_kernel<<<dim3(13, 13, B_), 512, 0, stream>>>(encp, decp, w2h, b2, targets,
                                                          blank_lp, emit_lp);
  alpha_kernel<<<B_, 64, 0, stream>>>(blank_lp, emit_lp, in_len, tg_len, ll);
  finish_kernel<<<1, 32, 0, stream>>>(ll, (float*)d_out);
}
